// SClusterFormer_12317966204958
// MI455X (gfx1250) — compile-verified, matrix operand order checked by reading, not run on hardware
//
#include <hip/hip_runtime.h>

// ---------------- problem constants (from reference) ----------------
#define D_MODEL 512
#define HH      192          // HEADS*HD
#define HH2     384          // fused feat|val width
#define HID     2048
#define NTOK    1024         // 32x32 tokens
#define BATCH   64
#define NHEADS  8
#define HD      24
#define MTOT    (BATCH * NTOK)   // 65536 rows

#ifndef USE_ASYNC_LDS
#define USE_ASYNC_LDS 1      // flip to 0 if global_load_async_to_lds asm fails
#endif

typedef __attribute__((ext_vector_type(16))) __bf16 bf16x16;
typedef __attribute__((ext_vector_type(8)))  __bf16 bf16x8;
typedef __attribute__((ext_vector_type(8)))  float  f32x8;

__device__ __forceinline__ float gelu_exact(float v) {
    return 0.5f * v * (1.0f + erff(v * 0.70710678118654752f));
}

#if USE_ASYNC_LDS
// LDS byte offset of a generic pointer into a __shared__ array.
__device__ __forceinline__ unsigned lds_off_of(const void* p) {
    return (unsigned)(unsigned long long)(__attribute__((address_space(3))) const void*)p;
}
// CDNA5 async copy global -> LDS (16B per lane), tracked with ASYNCcnt.
__device__ __forceinline__ void async_ld_b128(unsigned lds_off, const void* g) {
    unsigned long long ga = (unsigned long long)g;
    asm volatile("global_load_async_to_lds_b128 %0, %1, off"
                 :: "v"(lds_off), "v"(ga) : "memory");
}
__device__ __forceinline__ void wait_async0() {
    asm volatile("s_wait_asynccnt 0" ::: "memory");
}
#endif

// ---------------- f32 -> bf16 conversion (weights) ----------------
__global__ void cvt_f32_bf16(const float* __restrict__ in, __bf16* __restrict__ out, int n) {
    int i = blockIdx.x * blockDim.x + threadIdx.x;
    if (i < n) out[i] = (__bf16)in[i];
}

__global__ void concat_bias(const float* __restrict__ a, const float* __restrict__ b,
                            float* __restrict__ o, int n) {
    int i = blockIdx.x * blockDim.x + threadIdx.x;
    if (i < n) o[i] = a[i];
    else if (i < 2 * n) o[i] = b[i - n];
}

// ---------------- LayerNorm over 512, write bf16 ----------------
__global__ __launch_bounds__(256)
void ln512_bf16(const float* __restrict__ x, const float* __restrict__ g,
                const float* __restrict__ b, __bf16* __restrict__ out) {
    __shared__ float ssum[256];
    __shared__ float ssq[256];
    const int row = blockIdx.x;
    const int tid = threadIdx.x;
    const float* xr = x + (size_t)row * D_MODEL;
    float a0 = xr[tid], a1 = xr[tid + 256];
    ssum[tid] = a0 + a1;
    ssq[tid]  = a0 * a0 + a1 * a1;
    __syncthreads();
    for (int s = 128; s > 0; s >>= 1) {
        if (tid < s) { ssum[tid] += ssum[tid + s]; ssq[tid] += ssq[tid + s]; }
        __syncthreads();
    }
    float mean = ssum[0] * (1.0f / D_MODEL);
    float var  = ssq[0] * (1.0f / D_MODEL) - mean * mean;
    float inv  = rsqrtf(var + 1e-5f);
    __bf16* orow = out + (size_t)row * D_MODEL;
    orow[tid]       = (__bf16)((a0 - mean) * inv * g[tid]       + b[tid]);
    orow[tid + 256] = (__bf16)((a1 - mean) * inv * g[tid + 256] + b[tid + 256]);
}

// ---------------- bf16 WMMA GEMM: out = A[MxK] @ W[NxK]^T + bias (+res) ----------------
// Block tile 128(M) x 128(N), 8 waves in 4x2; each wave 32x64 via 2x4 WMMA frags.
// Double-buffered LDS staged with async global->LDS copies.
template<int GELU, int HAS_RES, typename OutT>
__global__ __launch_bounds__(256)
void gemm_bf16_wmma(const __bf16* __restrict__ A, const __bf16* __restrict__ W,
                    const float* __restrict__ bias, const float* __restrict__ res,
                    OutT* __restrict__ out, int M, int N, int K) {
    __shared__ __align__(16) __bf16 As[2][128 * 32];
    __shared__ __align__(16) __bf16 Ws[2][128 * 32];

    const int tid  = threadIdx.x;
    const int lane = tid & 31;
    const int wave = tid >> 5;
    const int wm   = wave & 3;   // 4 wave-rows of 32
    const int wn   = wave >> 2;  // 2 wave-cols of 64
    const int Mbase = blockIdx.x * 128;
    const int Nbase = blockIdx.y * 128;

    // 16-bit A/B fragment layout (ISA 7.12.2): lane<16 -> K[0..7],K[16..23];
    // lane>=16 -> K[8..15],K[24..31]; lane%16 selects the matrix row.
    const int kbase = (lane & 16) ? 8 : 0;
    const int mrow  = lane & 15;

    // per-thread staging chunks: tile is 128 rows x 32 cols = 512 x 16B chunks
    const int r0 = tid >> 2, part0 = tid & 3;            // chunk tid
    const int r1 = (tid + 256) >> 2, part1 = tid & 3;    // chunk tid+256

    f32x8 acc[2][4] = {};

    const int nk = K >> 5;  // K / 32

    // ---- staging helper (macro-free lambda) ----
    auto stage = [&](int buf, int k0) {
#if USE_ASYNC_LDS
        async_ld_b128(lds_off_of(&As[buf][r0 * 32 + part0 * 8]),
                      A + (size_t)(Mbase + r0) * K + k0 + part0 * 8);
        async_ld_b128(lds_off_of(&As[buf][r1 * 32 + part1 * 8]),
                      A + (size_t)(Mbase + r1) * K + k0 + part1 * 8);
        async_ld_b128(lds_off_of(&Ws[buf][r0 * 32 + part0 * 8]),
                      W + (size_t)(Nbase + r0) * K + k0 + part0 * 8);
        async_ld_b128(lds_off_of(&Ws[buf][r1 * 32 + part1 * 8]),
                      W + (size_t)(Nbase + r1) * K + k0 + part1 * 8);
#else
        *(float4*)&As[buf][r0 * 32 + part0 * 8] =
            *(const float4*)(A + (size_t)(Mbase + r0) * K + k0 + part0 * 8);
        *(float4*)&As[buf][r1 * 32 + part1 * 8] =
            *(const float4*)(A + (size_t)(Mbase + r1) * K + k0 + part1 * 8);
        *(float4*)&Ws[buf][r0 * 32 + part0 * 8] =
            *(const float4*)(W + (size_t)(Nbase + r0) * K + k0 + part0 * 8);
        *(float4*)&Ws[buf][r1 * 32 + part1 * 8] =
            *(const float4*)(W + (size_t)(Nbase + r1) * K + k0 + part1 * 8);
#endif
    };

    stage(0, 0);  // prologue

    for (int kb = 0; kb < nk; ++kb) {
        const int cur = kb & 1;
#if USE_ASYNC_LDS
        wait_async0();        // this wave's loads for 'cur' (issued last iter) done
#endif
        __syncthreads();      // all waves: 'cur' staged; prior reads of '!cur' done
        if (kb + 1 < nk) stage(cur ^ 1, (kb + 1) << 5);

        bf16x16 afrag[2], bfrag[4];
        #pragma unroll
        for (int i = 0; i < 2; ++i) {
            const __bf16* p = &As[cur][(wm * 32 + i * 16 + mrow) * 32];
            bf16x8 lo = *(const bf16x8*)(p + kbase);
            bf16x8 hi = *(const bf16x8*)(p + kbase + 16);
            #pragma unroll
            for (int e = 0; e < 8; ++e) { afrag[i][e] = lo[e]; afrag[i][8 + e] = hi[e]; }
        }
        #pragma unroll
        for (int j = 0; j < 4; ++j) {
            const __bf16* p = &Ws[cur][(wn * 64 + j * 16 + mrow) * 32];
            bf16x8 lo = *(const bf16x8*)(p + kbase);
            bf16x8 hi = *(const bf16x8*)(p + kbase + 16);
            #pragma unroll
            for (int e = 0; e < 8; ++e) { bfrag[j][e] = lo[e]; bfrag[j][8 + e] = hi[e]; }
        }
        #pragma unroll
        for (int i = 0; i < 2; ++i)
            #pragma unroll
            for (int j = 0; j < 4; ++j)
                acc[i][j] = __builtin_amdgcn_wmma_f32_16x16x32_bf16(
                    false, afrag[i], false, bfrag[j], (short)0, acc[i][j], false, false);
    }

    // Epilogue. C/D layout: lanes 0-15 hold rows 0..7 (vgpr idx), lanes 16-31 rows 8..15.
    const int col0  = lane & 15;
    const int rhalf = (lane >> 4) * 8;
    #pragma unroll
    for (int i = 0; i < 2; ++i) {
        #pragma unroll
        for (int j = 0; j < 4; ++j) {
            int gc = Nbase + wn * 64 + j * 16 + col0;
            float bv = bias[gc];
            #pragma unroll
            for (int r = 0; r < 8; ++r) {
                int gr = Mbase + wm * 32 + i * 16 + rhalf + r;
                float v = acc[i][j][r] + bv;
                if (GELU) v = gelu_exact(v);
                size_t idx = (size_t)gr * N + gc;
                if (HAS_RES) v += res[idx];
                out[idx] = (OutT)v;
            }
        }
    }
}

// ---------------- Cluster3D core: one workgroup per (batch, head) ----------------
// fv holds fused [feat | val] rows of width HH2; feat at +h*HD, val at +HH+h*HD.
__global__ __launch_bounds__(256)
void cluster3d_kernel(const float* __restrict__ fv,
                      const float* __restrict__ alpha_p, const float* __restrict__ beta_p,
                      __bf16* __restrict__ outc) {
    const int h = blockIdx.x, b = blockIdx.y;
    const int tid = threadIdx.x;
    const size_t fbase = (size_t)b * NTOK * HH2 + (size_t)h * HD;        // feat
    const size_t vbase = fbase + HH;                                     // val
    const size_t obase = (size_t)b * NTOK * HH + (size_t)h * HD;         // output

    __shared__ float cn[16][HD];     // centers (then normalized)
    __shared__ float vcent[16][HD];  // pooled value centers
    __shared__ float agg[16][HD];
    __shared__ float cnt[16];
    __shared__ float cnorm[16];
    __shared__ float simv[NTOK];
    __shared__ int   sidx[NTOK];

    // Phase 1: adaptive avg pool 32x32 -> 4x4 (each center pools an 8x8 token block)
    for (int p = tid; p < 16 * HD; p += 256) {
        int m = p / HD, d = p % HD;
        int p1 = m >> 2, p2 = m & 3;
        float sf = 0.f, sv = 0.f;
        for (int r = 0; r < 8; ++r)
            for (int c = 0; c < 8; ++c) {
                int n = (p1 * 8 + r) * 32 + (p2 * 8 + c);
                sf += fv[fbase + (size_t)n * HH2 + d];
                sv += fv[vbase + (size_t)n * HH2 + d];
            }
        cn[m][d]    = sf * (1.0f / 64.0f);
        vcent[m][d] = sv * (1.0f / 64.0f);
        agg[m][d]   = 0.f;
    }
    if (tid < 16) cnt[tid] = 0.f;
    __syncthreads();
    if (tid < 16) {
        float s = 0.f;
        for (int d = 0; d < HD; ++d) s += cn[tid][d] * cn[tid][d];
        cnorm[tid] = fmaxf(sqrtf(s), 1e-12f);
    }
    __syncthreads();
    for (int p = tid; p < 16 * HD; p += 256) {
        int m = p / HD, d = p % HD;
        cn[m][d] /= cnorm[m];
    }
    __syncthreads();

    const float alpha = alpha_p[0];
    const float beta  = beta_p[0];

    // Phase 2: per-token similarity, LeakyReLU, hard top-1, LDS-atomic aggregation
    for (int n = tid; n < NTOK; n += 256) {
        float f[HD], v[HD];
        float nrm = 0.f;
        #pragma unroll
        for (int d = 0; d < HD; ++d) {
            f[d] = fv[fbase + (size_t)n * HH2 + d];
            v[d] = fv[vbase + (size_t)n * HH2 + d];
            nrm += f[d] * f[d];
        }
        float inv = 1.0f / fmaxf(sqrtf(nrm), 1e-12f);
        float best = -3.0e38f; int bi = 0;
        for (int m = 0; m < 16; ++m) {
            float s = 0.f;
            #pragma unroll
            for (int d = 0; d < HD; ++d) s += cn[m][d] * f[d];
            s = beta + alpha * s * inv;
            s = (s >= 0.f) ? s : 0.2f * s;   // LeakyReLU(0.2)
            if (s > best) { best = s; bi = m; }   // first-max tie-break like argmax
        }
        simv[n] = best;
        sidx[n] = bi;
        atomicAdd(&cnt[bi], 1.0f);
        #pragma unroll
        for (int d = 0; d < HD; ++d) atomicAdd(&agg[bi][d], best * v[d]);  // ds_add_f32
    }
    __syncthreads();

    // Phase 3: agg = (sim@v + vcenters) / (count + 1)
    for (int p = tid; p < 16 * HD; p += 256) {
        int m = p / HD, d = p % HD;
        agg[m][d] = (agg[m][d] + vcent[m][d]) / (cnt[m] + 1.0f);
    }
    __syncthreads();

    // Phase 4: dispatch back, bf16 for the proj GEMM
    for (int n = tid; n < NTOK; n += 256) {
        float s = simv[n]; int m = sidx[n];
        #pragma unroll
        for (int d = 0; d < HD; ++d)
            outc[obase + (size_t)n * HH + d] = (__bf16)(s * agg[m][d]);
    }
}

// ---------------- host-side orchestration ----------------
extern "C" void kernel_launch(void* const* d_in, const int* in_sizes, int n_in,
                              void* d_out, int out_size, void* d_ws, size_t ws_size,
                              hipStream_t stream) {
    (void)in_sizes; (void)n_in; (void)out_size; (void)ws_size;

    const float* x     = (const float*)d_in[0];
    const float* ln1_g = (const float*)d_in[1];
    const float* ln1_b = (const float*)d_in[2];
    const float* f_w   = (const float*)d_in[3];
    const float* f_b   = (const float*)d_in[4];
    const float* v_w   = (const float*)d_in[5];
    const float* v_b   = (const float*)d_in[6];
    const float* alpha = (const float*)d_in[7];
    const float* beta  = (const float*)d_in[8];
    const float* p_w   = (const float*)d_in[9];
    const float* p_b   = (const float*)d_in[10];
    const float* ln2_g = (const float*)d_in[11];
    const float* ln2_b = (const float*)d_in[12];
    const float* w1    = (const float*)d_in[13];
    const float* b1    = (const float*)d_in[14];
    const float* w2    = (const float*)d_in[15];
    const float* b2    = (const float*)d_in[16];
    float* out = (float*)d_out;

    // carve workspace
    char* p = (char*)d_ws;
    auto carve = [&](size_t bytes) -> char* {
        char* r = p;
        p += (bytes + 255) & ~(size_t)255;
        return r;
    };
    __bf16* xn   = (__bf16*)carve((size_t)MTOT * D_MODEL * 2);  // LN1(x) bf16
    float*  fv   = (float*) carve((size_t)MTOT * HH2 * 4);      // fused feat|val
    __bf16* clus = (__bf16*)carve((size_t)MTOT * HH * 2);       // cluster output bf16
    float*  x1   = (float*) carve((size_t)MTOT * D_MODEL * 4);  // residual after proj
    __bf16* h2   = (__bf16*)carve((size_t)MTOT * D_MODEL * 2);  // LN2(x1) bf16
    __bf16* hbuf = (__bf16*)carve((size_t)MTOT * HID * 2);      // GELU(fc1) bf16
    __bf16* wfv  = (__bf16*)carve((size_t)HH2 * D_MODEL * 2);   // [f_w ; v_w] bf16
    float*  bfv  = (float*) carve((size_t)HH2 * 4);             // [f_b ; v_b]
    __bf16* pwb  = (__bf16*)carve((size_t)D_MODEL * HH * 2);
    __bf16* w1b  = (__bf16*)carve((size_t)HID * D_MODEL * 2);
    __bf16* w2b  = (__bf16*)carve((size_t)D_MODEL * HID * 2);

    // weight conversions f32 -> bf16 (v_w lands directly after f_w -> fused [384 x 512])
    cvt_f32_bf16<<<(HH * D_MODEL + 255) / 256, 256, 0, stream>>>(f_w, wfv, HH * D_MODEL);
    cvt_f32_bf16<<<(HH * D_MODEL + 255) / 256, 256, 0, stream>>>(v_w, wfv + (size_t)HH * D_MODEL,
                                                                 HH * D_MODEL);
    concat_bias<<<(HH2 + 255) / 256, 256, 0, stream>>>(f_b, v_b, bfv, HH);
    cvt_f32_bf16<<<(D_MODEL * HH + 255) / 256, 256, 0, stream>>>(p_w, pwb, D_MODEL * HH);
    cvt_f32_bf16<<<(HID * D_MODEL + 255) / 256, 256, 0, stream>>>(w1, w1b, HID * D_MODEL);
    cvt_f32_bf16<<<(D_MODEL * HID + 255) / 256, 256, 0, stream>>>(w2, w2b, D_MODEL * HID);

    // LN1
    ln512_bf16<<<MTOT, 256, 0, stream>>>(x, ln1_g, ln1_b, xn);

    // fused feat|val GEMM: [65536 x 512] @ [512 x 384] (single pass over xn)
    gemm_bf16_wmma<0, 0, float><<<dim3(MTOT / 128, HH2 / 128), 256, 0, stream>>>(
        xn, wfv, bfv, nullptr, fv, MTOT, HH2, D_MODEL);

    // clustering (per batch x head)
    cluster3d_kernel<<<dim3(NHEADS, BATCH), 256, 0, stream>>>(fv, alpha, beta, clus);

    // proj + residual: x1 = x + clus @ proj_w^T + proj_b
    gemm_bf16_wmma<0, 1, float><<<dim3(MTOT / 128, D_MODEL / 128), 256, 0, stream>>>(
        clus, pwb, p_b, x, x1, MTOT, D_MODEL, HH);

    // LN2
    ln512_bf16<<<MTOT, 256, 0, stream>>>(x1, ln2_g, ln2_b, h2);

    // fc1 + exact GELU (bf16 out)
    gemm_bf16_wmma<1, 0, __bf16><<<dim3(MTOT / 128, HID / 128), 256, 0, stream>>>(
        h2, w1b, b1, nullptr, hbuf, MTOT, HID, D_MODEL);

    // fc2 + residual -> final f32 output
    gemm_bf16_wmma<0, 1, float><<<dim3(MTOT / 128, D_MODEL / 128), 256, 0, stream>>>(
        hbuf, w2b, b2, x1, out, MTOT, D_MODEL, HID);
}
